// Attention_81793357185069
// MI455X (gfx1250) — matrix operand order repaired and checked by reading, not compiled, not run on hardware
//
#include <hip/hip_runtime.h>
#include <hip/hip_bf16.h>
#include <math.h>

// ---------------------------------------------------------------------------
// Types for CDNA5 WMMA (wave32): 16x16x32 f16 -> f32, plus TDM descriptors
// ---------------------------------------------------------------------------
typedef _Float16 h16;
typedef __attribute__((ext_vector_type(16))) _Float16 v16h;
typedef __attribute__((ext_vector_type(8)))  _Float16 v8h;
typedef __attribute__((ext_vector_type(8)))  float    v8f;
typedef __attribute__((ext_vector_type(4)))  unsigned int v4u;
typedef __attribute__((ext_vector_type(8)))  int      v8i;
typedef __attribute__((ext_vector_type(4)))  int      v4i;

#define N_BATCH 16
#define SEQ_L   1024
#define TOKENS  (N_BATCH * SEQ_L)   // 16384
#define DMODEL  512
#define GROUPS  4
#define GP      128                 // planes per group
#define ODIM    2048

union AFrag { v16h v; v8h h[2]; };

// A-fragment (16xK tile, rows = lane%16) from a row-major f16 matrix.
// ISA layout: lane<16 -> halves {K 0..7, 16..23}; lane>=16 -> {8..15, 24..31}.
__device__ __forceinline__ v16h load_afrag(const h16* base, int ld, int row0, int k0) {
  int lane = threadIdx.x & 31;
  int m = lane & 15, hh = lane >> 4;
  const h16* p = base + (size_t)(row0 + m) * ld + k0 + hh * 8;
  AFrag u;
  u.h[0] = *(const v8h*)p;          // K = h*8 .. h*8+7
  u.h[1] = *(const v8h*)(p + 16);   // K = 16+h*8 .. 16+h*8+7
  return u.v;
}

// Same, but source is f32 (converted on the fly) — used for raw activations.
__device__ __forceinline__ v16h load_afrag_f32(const float* base, int ld, int row0, int k0) {
  int lane = threadIdx.x & 31;
  int m = lane & 15, hh = lane >> 4;
  const float* p = base + (size_t)(row0 + m) * ld + k0 + hh * 8;
  v8f x = *(const v8f*)p;
  v8f y = *(const v8f*)(p + 16);
  v16h r;
#pragma unroll
  for (int i = 0; i < 8; i++) { r[i] = (_Float16)x[i]; r[8 + i] = (_Float16)y[i]; }
  return r;
}

// B-fragment (Kx16 tile). For D = X*W^T, B column n is row (col0+n) of row-major W:
// same K order as A: lane<16 -> {K 0..7, 16..23}; lane>=16 -> {8..15, 24..31}.
__device__ __forceinline__ v16h load_bfrag(const h16* base, int ld, int col0, int k0) {
  int lane = threadIdx.x & 31;
  int n = lane & 15, hh = lane >> 4;
  const h16* p = base + (size_t)(col0 + n) * ld + k0 + hh * 8;
  AFrag u;
  u.h[0] = *(const v8h*)p;
  u.h[1] = *(const v8h*)(p + 16);
  return u.v;
}

__device__ __forceinline__ v8f wmma16(v16h a, v16h b, v8f c) {
  return __builtin_amdgcn_wmma_f32_16x16x32_f16(false, a, false, b, (short)0, c, false, false);
}

// ---------------------------------------------------------------------------
// TDM: stage a contiguous `bytes` block (multiple of 8) from global into LDS
// offset 0 of this workgroup. Must be issued by one wave; caller barriers.
// D# built per CDNA5 ISA §8.3-8.6: count=1, type=2, data_size=8B, 1-line tile.
// This toolchain exposes the 6-arg builtin:
//   (uint32x4 g0, int32x8 g1, int32x4 g2, int32x4 g3, int32x8 extra, i32 cpol)
// ---------------------------------------------------------------------------
__device__ __forceinline__ void tdm_copy_to_lds(const void* gsrc, unsigned bytes) {
  unsigned long long ga = (unsigned long long)gsrc;
  unsigned n8 = bytes >> 3;                 // elements of 8 bytes
  v4u g0;
  g0[0] = 1u;                               // count=1, user mode, no gather
  g0[1] = 0u;                               // lds_addr = 0 (workgroup base)
  g0[2] = (unsigned)(ga & 0xffffffffu);     // global_addr[31:0]
  g0[3] = (unsigned)((ga >> 32) & 0x1ffffffu) | (2u << 30);  // addr[56:32] | type=2
  v8i g1;
  g1[0] = (int)(3u << 16);                  // workgroup_mask=0, data_size=3 (8B)
  g1[1] = (int)((n8 & 0xffffu) << 16);      // tensor_dim0[15:0]
  g1[2] = (int)(((n8 >> 16) & 0xffffu) | (1u << 16)); // tensor_dim0[31:16] | tensor_dim1=1
  g1[3] = (int)((n8 & 0xffffu) << 16);      // tensor_dim1 hi=0 | tile_dim0=n8
  g1[4] = 1;                                // tile_dim1=1, tile_dim2 unused
  g1[5] = (int)n8;                          // tensor_dim0_stride[31:0]
  g1[6] = (int)((n8 & 0xffffu) << 16);      // stride hi=0 | tensor_dim1_stride[15:0]
  g1[7] = 0;                                // tensor_dim1_stride hi
  v4i g2;
  g2[0] = 1;                                // tensor_dim2=1
  g2[1] = 1;                                // tensor_dim3=1
  g2[2] = (int)n8;                          // tensor_dim2_stride lo
  g2[3] = 0;                                // stride hi | tile_dim3=0
  v4i g3;
  g3[0] = (int)n8;                          // tensor_dim3_stride lo
  g3[1] = (int)(1u << 16);                  // stride hi=0 | tensor_dim4=1
  g3[2] = 0;                                // tensor_dim4 hi | tile_dim4=0
  g3[3] = 0;
  v8i gx;
  gx[0] = 0; gx[1] = 0; gx[2] = 0; gx[3] = 0;
  gx[4] = 0; gx[5] = 0; gx[6] = 0; gx[7] = 0;
  __builtin_amdgcn_tensor_load_to_lds(g0, g1, g2, g3, gx, 0);
  __builtin_amdgcn_s_wait_tensorcnt((short)0);
}

// ---------------------------------------------------------------------------
// Kernel 0: f32 -> f16 weight conversion with folded EqualLinear scale
// ---------------------------------------------------------------------------
__global__ void k_cvt(const float* __restrict__ in, h16* __restrict__ out,
                      float scale, int n) {
  int i = blockIdx.x * blockDim.x + threadIdx.x;
  int stride = gridDim.x * blockDim.x;
  for (; i < n; i += stride) out[i] = (h16)(in[i] * scale);
}

// ---------------------------------------------------------------------------
// Kernel 1: QKV projections. blockIdx.z: 0=Q, 1=K, 2=V.
// Q,K stored token-major (TOKENS x 512); V stored plane-major (N,512,L)
// so it directly serves as B-fragments in the SV GEMM.
// Wave tile: 16 rows x 64 cols, K = 512 in 16 chunks of 32.
// ---------------------------------------------------------------------------
__global__ __launch_bounds__(256) void k_qkv(
    const float* __restrict__ par, const float* __restrict__ att,
    const h16* __restrict__ qw, const h16* __restrict__ kw, const h16* __restrict__ vw,
    const float* __restrict__ qb, const float* __restrict__ kb, const float* __restrict__ vb,
    h16* __restrict__ qo, h16* __restrict__ ko, h16* __restrict__ vo) {
  const int mode = blockIdx.z;
  const float* src  = (mode == 0) ? par : att;
  const h16*   W    = (mode == 0) ? qw : ((mode == 1) ? kw : vw);
  const float* bias = (mode == 0) ? qb : ((mode == 1) ? kb : vb);

  int w   = blockIdx.x * 8 + (threadIdx.x >> 5);  // 8192 waves
  int rt  = w >> 3;                               // row tile 0..1023
  int ct  = w & 7;                                // col tile 0..7
  int row0 = rt * 16, col0 = ct * 64;

  v8f acc[4] = {};
#pragma unroll 1
  for (int k0 = 0; k0 < DMODEL; k0 += 32) {
    v16h a = load_afrag_f32(src, DMODEL, row0, k0);
#pragma unroll
    for (int j = 0; j < 4; j++) {
      v16h b = load_bfrag(W, DMODEL, col0 + j * 16, k0);
      acc[j] = wmma16(a, b, acc[j]);
    }
  }

  int lane = threadIdx.x & 31;
  int cn = lane & 15, hh = lane >> 4;
#pragma unroll
  for (int j = 0; j < 4; j++) {
    int col = col0 + j * 16 + cn;
    float bb = bias[col];
#pragma unroll
    for (int r = 0; r < 8; r++) {
      int row = row0 + r + 8 * hh;                // D layout: (r + 8h, lane&15)
      float val = acc[j][r] + bb;
      if (mode < 2) {
        h16* o = (mode == 0) ? qo : ko;
        o[(size_t)row * DMODEL + col] = (h16)val;
      } else {
        int nb = row >> 10, l = row & 1023;       // transpose: plane-major V
        vo[((size_t)nb * DMODEL + col) * SEQ_L + l] = (h16)val;
      }
    }
  }
}

// ---------------------------------------------------------------------------
// Kernel 2: flash attention. One wave per block (barriers -> NOP), one
// (n, g, 16-row m-block) per wave. Online softmax; P transposed via LDS.
// Prefetches next K/V tiles one step ahead (global_prefetch_b8).
// ---------------------------------------------------------------------------
__global__ __launch_bounds__(32) void k_attn(
    const h16* __restrict__ qh, const h16* __restrict__ kh,
    const h16* __restrict__ vh, h16* __restrict__ svh) {
  __shared__ h16 ldsP[16 * 32];

  int b  = blockIdx.x;          // N * GROUPS * 64 = 4096
  int n  = b >> 8;
  int g  = (b >> 6) & 3;
  int mt = b & 63;
  int m0 = mt * 16;

  int lane = threadIdx.x & 31;
  int cn = lane & 15, hh = lane >> 4;

  const h16* qbase = qh + (size_t)n * SEQ_L * DMODEL;
  const h16* kbase = kh + (size_t)n * SEQ_L * DMODEL;
  const h16* vbase = vh + ((size_t)n * DMODEL + g * GP) * SEQ_L;  // plane-major

  v16h qf[4];
#pragma unroll
  for (int c = 0; c < 4; c++) qf[c] = load_afrag(qbase, DMODEL, m0, g * GP + c * 32);

  v8f o[8] = {};
  float mrun[8], srun[8];
#pragma unroll
  for (int r = 0; r < 8; r++) { mrun[r] = -3.0e38f; srun[r] = 0.f; }

  const float ascale = 0.044194173824159216f;  // PLANES^-0.5 = 512^-0.5

#pragma unroll 1
  for (int l0 = 0; l0 < SEQ_L; l0 += 32) {
    // ---- prefetch next l-step's K rows and V row segments ----
    if (l0 + 32 < SEQ_L) {
      __builtin_prefetch(kbase + (size_t)(l0 + 32 + lane) * DMODEL + g * GP, 0, 0);
#pragma unroll
      for (int r = 0; r < 4; r++)
        __builtin_prefetch(vbase + (size_t)(lane + r * 32) * SEQ_L + l0 + 32, 0, 0);
    }
    // ---- scores: two 16x16 tiles over columns [l0, l0+32) ----
    v8f s0 = {}, s1 = {};
#pragma unroll
    for (int c = 0; c < 4; c++) {
      v16h b0 = load_bfrag(kbase, DMODEL, l0,      g * GP + c * 32);
      v16h b1 = load_bfrag(kbase, DMODEL, l0 + 16, g * GP + c * 32);
      s0 = wmma16(qf[c], b0, s0);
      s1 = wmma16(qf[c], b1, s1);
    }
    // ---- online softmax (row = r + 8*hh; row spread over 16 lanes of half) ----
    float corr[8];
#pragma unroll
    for (int r = 0; r < 8; r++) {
      float a0 = s0[r] * ascale, a1 = s1[r] * ascale;
      float t = fmaxf(a0, a1);
      t = fmaxf(t, __shfl_xor(t, 1, 32));
      t = fmaxf(t, __shfl_xor(t, 2, 32));
      t = fmaxf(t, __shfl_xor(t, 4, 32));
      t = fmaxf(t, __shfl_xor(t, 8, 32));
      float mn = fmaxf(mrun[r], t);
      float c0 = __expf(mrun[r] - mn);
      float p0 = __expf(a0 - mn), p1 = __expf(a1 - mn);
      float rs = p0 + p1;
      rs += __shfl_xor(rs, 1, 32);
      rs += __shfl_xor(rs, 2, 32);
      rs += __shfl_xor(rs, 4, 32);
      rs += __shfl_xor(rs, 8, 32);
      srun[r] = srun[r] * c0 + rs;
      mrun[r] = mn;
      corr[r] = c0;
      s0[r] = p0; s1[r] = p1;
    }
#pragma unroll
    for (int t = 0; t < 8; t++)
#pragma unroll
      for (int r = 0; r < 8; r++) o[t][r] *= corr[r];

    // ---- transpose P (D layout -> A fragment) through LDS ----
#pragma unroll
    for (int r = 0; r < 8; r++) {
      ldsP[(r + 8 * hh) * 32 + cn]      = (h16)s0[r];
      ldsP[(r + 8 * hh) * 32 + 16 + cn] = (h16)s1[r];
    }
    asm volatile("s_wait_dscnt 0" ::: "memory");
    v16h pa = load_afrag(ldsP, 32, 0, 0);
    asm volatile("" ::: "memory");

    // ---- O += P(16x32) x V(32x128): 8 d-tiles ----
#pragma unroll
    for (int t = 0; t < 8; t++) {
      v16h vb = load_bfrag(vbase, SEQ_L, t * 16, l0);
      o[t] = wmma16(pa, vb, o[t]);
    }
  }

  // ---- finalize: divide by softmax denom, store sv token-major (for proj A) ----
  h16* sbase = svh + ((size_t)n * SEQ_L + m0) * DMODEL + g * GP;
#pragma unroll
  for (int t = 0; t < 8; t++)
#pragma unroll
    for (int r = 0; r < 8; r++) {
      float val = o[t][r] / srun[r];
      sbase[(size_t)(r + 8 * hh) * DMODEL + t * 16 + cn] = (h16)val;
    }
}

// ---------------------------------------------------------------------------
// Kernel 3: output projection 16384 x 2048, K=512; f32 out + bias to d_out.
// All 8 waves of a block share one 16x512 A tile: staged ONCE into LDS by the
// Tensor Data Mover (tensor_load_to_lds + s_wait_tensorcnt), then read as
// ds_load_b128 A-fragments. Block covers col tiles [ (b&3)*8 .. +8 ).
// ---------------------------------------------------------------------------
__global__ __launch_bounds__(256) void k_proj(
    const h16* __restrict__ a, const h16* __restrict__ w,
    const float* __restrict__ bias, float* __restrict__ out) {
  __shared__ __align__(16) h16 ldsA[16 * DMODEL];   // 16 KB

  int b  = blockIdx.x;                  // 4096 blocks
  int rt = b >> 2;                      // 0..1023
  int ct = (b & 3) * 8 + (threadIdx.x >> 5);  // 0..31
  int row0 = rt * 16, col0 = ct * 64;

  if ((threadIdx.x >> 5) == 0) {
    tdm_copy_to_lds(a + (size_t)row0 * DMODEL, 16 * DMODEL * sizeof(h16));
  }
  __syncthreads();

  v8f acc[4] = {};
#pragma unroll 1
  for (int k0 = 0; k0 < DMODEL; k0 += 32) {
    v16h av = load_afrag(ldsA, DMODEL, 0, k0);
#pragma unroll
    for (int j = 0; j < 4; j++) {
      v16h bv = load_bfrag(w, DMODEL, col0 + j * 16, k0);
      acc[j] = wmma16(av, bv, acc[j]);
    }
  }

  int lane = threadIdx.x & 31;
  int cn = lane & 15, hh = lane >> 4;
#pragma unroll
  for (int j = 0; j < 4; j++) {
    int col = col0 + j * 16 + cn;
    float bb = bias[col];
#pragma unroll
    for (int r = 0; r < 8; r++) {
      int row = row0 + r + 8 * hh;
      out[(size_t)row * ODIM + col] = acc[j][r] + bb;
    }
  }
}

// ---------------------------------------------------------------------------
// Host-side launch
// ---------------------------------------------------------------------------
extern "C" void kernel_launch(void* const* d_in, const int* in_sizes, int n_in,
                              void* d_out, int out_size, void* d_ws, size_t ws_size,
                              hipStream_t stream) {
  (void)in_sizes; (void)n_in; (void)out_size; (void)ws_size;
  const float* attention = (const float*)d_in[0];
  const float* op_param  = (const float*)d_in[1];
  const float* q_w = (const float*)d_in[2];
  const float* q_b = (const float*)d_in[3];
  const float* k_w = (const float*)d_in[4];
  const float* k_b = (const float*)d_in[5];
  const float* v_w = (const float*)d_in[6];
  const float* v_b = (const float*)d_in[7];
  const float* proj_w = (const float*)d_in[8];
  const float* proj_b = (const float*)d_in[9];
  float* out = (float*)d_out;
  h16* ws = (h16*)d_ws;

  // workspace layout (element = f16); total ~35.4M halves = ~70.8 MB
  size_t o_qw = 0;
  size_t o_kw = o_qw + (size_t)512 * 512;
  size_t o_vw = o_kw + (size_t)512 * 512;
  size_t o_pw = o_vw + (size_t)512 * 512;
  size_t o_q  = o_pw + (size_t)2048 * 512;
  size_t o_k  = o_q  + (size_t)TOKENS * 512;
  size_t o_v  = o_k  + (size_t)TOKENS * 512;
  size_t o_sv = o_v  + (size_t)TOKENS * 512;

  const float wscale = 0.044194173824159216f;  // 1/sqrt(512)

  k_cvt<<<256, 256, 0, stream>>>(q_w,    ws + o_qw, wscale, 512 * 512);
  k_cvt<<<256, 256, 0, stream>>>(k_w,    ws + o_kw, wscale, 512 * 512);
  k_cvt<<<256, 256, 0, stream>>>(v_w,    ws + o_vw, wscale, 512 * 512);
  k_cvt<<<1024, 256, 0, stream>>>(proj_w, ws + o_pw, wscale, 2048 * 512);

  k_qkv<<<dim3(1024, 1, 3), 256, 0, stream>>>(
      op_param, attention,
      ws + o_qw, ws + o_kw, ws + o_vw,
      q_b, k_b, v_b,
      ws + o_q, ws + o_k, ws + o_v);

  k_attn<<<4096, 32, 0, stream>>>(ws + o_q, ws + o_k, ws + o_v, ws + o_sv);

  k_proj<<<4096, 256, 0, stream>>>(ws + o_sv, ws + o_pw, proj_b, out);
}
